// SelfAttentionBlock_16578573762631
// MI455X (gfx1250) — matrix operand order repaired and checked by reading, not compiled, not run on hardware
//
#include <hip/hip_runtime.h>
#include <hip/hip_bf16.h>

// ---------------------------------------------------------------------------
// SAGAN self-attention block, fused flash-attention style for gfx1250 (MI455X)
//   f = Wf x + bf   (B,32,N)   -> stored bf16 as fT (B,N,32)
//   g = Wg x + bg   (B,32,N)   -> stored bf16 as gT (B,N,32)
//   h = Wh x + bh   (B,256,N)  -> stored bf16 as hB (B,256,N)
//   s = f^T g ; att = softmax(s); out = gamma * (h att^T) + x
// Score matrix never materialized (online softmax). Matmuls on
// v_wmma_f32_16x16x32_bf16; K/V tiles staged with global->LDS async copies
// (ASYNCcnt) via __builtin_amdgcn_global_load_async_to_lds_b128.
// ---------------------------------------------------------------------------

typedef __bf16 bf16_t;
typedef __attribute__((ext_vector_type(16))) __bf16 v16bf;
typedef __attribute__((ext_vector_type(8)))  float  v8f;

// async global->LDS builtin operand types: int4 pointers in explicit
// address spaces (addrspace(1) = global/device, addrspace(3) = LDS).
typedef int v4i_t __attribute__((vector_size(4 * sizeof(int))));
typedef __attribute__((address_space(1))) v4i_t gv4i_t;
typedef __attribute__((address_space(3))) v4i_t lv4i_t;

constexpr int Bc  = 4;     // batch
constexpr int Cc  = 256;   // channels
constexpr int CKc = 32;    // qk projection dim
constexpr int Nc  = 4096;  // pixels (64*64)

constexpr int QPG    = 16; // queries per group (one 16-row tile)
constexpr int GROUPS = 4;  // query groups per block (2 waves each -> 8 waves)
constexpr int KT     = 64; // keys per main-loop iteration (2 x WMMA K)
// LDS row strides (bf16 units). Chosen so (a) async B128 destinations are
// 16B-aligned (stride % 8 == 0 where async-staged) and (b) the 16 lanes of a
// WMMA fragment read hit 16 distinct banks.
constexpr int VSTR = 72;   // sV: dword stride 36 -> banks {36c%64} distinct
constexpr int KSTR = 40;   // sK: dword stride 20 -> banks {20j%64} distinct
constexpr int PSTR = 66;   // sP: dword stride 33 (odd) -> distinct banks

#if defined(__gfx1250__) &&                                          \
    __has_builtin(__builtin_amdgcn_global_load_async_to_lds_b128) && \
    __has_builtin(__builtin_amdgcn_s_wait_asynccnt)
#define USE_ASYNC_LDS 1
#else
#define USE_ASYNC_LDS 0
#endif

// ---------------------------------------------------------------------------
// Projection: f and g (32 output channels each), one thread per pixel.
// ---------------------------------------------------------------------------
__global__ __launch_bounds__(256) void proj_fg(
    const float* __restrict__ x, const float* __restrict__ Wf,
    const float* __restrict__ bf, const float* __restrict__ Wg,
    const float* __restrict__ bg, bf16_t* __restrict__ fT,
    bf16_t* __restrict__ gT) {
  int gid = blockIdx.x * blockDim.x + threadIdx.x;  // 0 .. B*N-1
  int b = gid / Nc;
  int n = gid - b * Nc;
  const float* xp = x + (size_t)b * Cc * Nc + n;
  float accF[CKc], accG[CKc];
#pragma unroll
  for (int k = 0; k < CKc; ++k) { accF[k] = bf[k]; accG[k] = bg[k]; }
  for (int c = 0; c < Cc; ++c) {
    float xv = xp[(size_t)c * Nc];
#pragma unroll
    for (int k = 0; k < CKc; ++k) {
      accF[k] = fmaf(Wf[k * Cc + c], xv, accF[k]);
      accG[k] = fmaf(Wg[k * Cc + c], xv, accG[k]);
    }
  }
  bf16_t* fo = fT + (size_t)gid * CKc;
  bf16_t* go = gT + (size_t)gid * CKc;
#pragma unroll
  for (int k = 0; k < CKc; ++k) { fo[k] = (bf16_t)accF[k]; go[k] = (bf16_t)accG[k]; }
}

// ---------------------------------------------------------------------------
// Projection: h (256 output channels). Block = (b, c, 256-pixel chunk).
// ---------------------------------------------------------------------------
__global__ __launch_bounds__(256) void proj_h(
    const float* __restrict__ x, const float* __restrict__ Wh,
    const float* __restrict__ bh, bf16_t* __restrict__ hB) {
  int nchunks = Nc / 256;
  int n = (blockIdx.x % nchunks) * 256 + threadIdx.x;
  int c = (blockIdx.x / nchunks) % Cc;
  int b = blockIdx.x / (nchunks * Cc);
  const float* xp = x + (size_t)b * Cc * Nc + n;
  const float* wr = Wh + (size_t)c * Cc;
  float acc = bh[c];
  for (int c2 = 0; c2 < Cc; ++c2) acc = fmaf(wr[c2], xp[(size_t)c2 * Nc], acc);
  hB[((size_t)b * Cc + c) * Nc + n] = (bf16_t)acc;
}

// ---------------------------------------------------------------------------
// Fused attention. Block = 256 threads = 8 waves.
//   wave w: query group qg = w/2 (16 queries), channel half = w%2 (128 chans)
// Per 64-key tile: 4 QK WMMAs, online softmax (stats amortized over 64 keys),
// P -> bf16 A-fragments via LDS bounce, O += P x V via 16 WMMAs.
// ---------------------------------------------------------------------------
__global__ __launch_bounds__(256) void attn(
    const bf16_t* __restrict__ fT, const bf16_t* __restrict__ gT,
    const bf16_t* __restrict__ hB, const float* __restrict__ x,
    const float* __restrict__ gamma, float* __restrict__ out) {
  __shared__ bf16_t sV[Cc * VSTR];                 // [channel][key]   V tile
  __shared__ bf16_t sK[KT * KSTR];                 // [key][qk-chan]   K tile
  __shared__ bf16_t sP[GROUPS * QPG * PSTR];       // [group][row][key] P bounce

  const int tid   = threadIdx.x;
  const int wave  = tid >> 5;
  const int lane  = tid & 31;
  const int qg    = wave >> 1;      // 0..3
  const int chalf = wave & 1;       // 0..1
  const int half  = lane >> 4;      // hi/lo 16-lane half of the wave
  const int l15   = lane & 15;

  const int qblocks = Nc / (GROUPS * QPG);         // 64 query blocks per batch
  const int b  = blockIdx.x / qblocks;
  const int q0 = (blockIdx.x % qblocks) * (GROUPS * QPG) + qg * QPG;

  // --- Q fragment, A layout: lane = M (query row), elems split per half ---
  v16bf aq;
  {
    const bf16_t* qp = fT + ((size_t)b * Nc + (q0 + l15)) * CKc;
#pragma unroll
    for (int e = 0; e < 16; ++e) {
      int k = e + 16 * half;                       // A layout: K=e+16*half (matches B)
      aq[e] = qp[k];
    }
  }

  v8f o[8];                                        // 16q x 128ch accumulator
#pragma unroll
  for (int t = 0; t < 8; ++t)
#pragma unroll
    for (int r = 0; r < 8; ++r) o[t][r] = 0.f;

  float mrow[8], lrow[8];                          // online softmax stats
#pragma unroll
  for (int r = 0; r < 8; ++r) { mrow[r] = -1e30f; lrow[r] = 0.f; }

  const v8f zero8 = {0.f, 0.f, 0.f, 0.f, 0.f, 0.f, 0.f, 0.f};

  for (int j0 = 0; j0 < Nc; j0 += KT) {
    __syncthreads();  // WAR: previous iteration's LDS reads done block-wide

#if USE_ASYNC_LDS
    // ---- async global->LDS staging (ASYNCcnt path) ----
    {
      // V tile: 256 channels x 64 keys; each thread DMAs its 128B row.
      const bf16_t* hp = hB + ((size_t)b * Cc + tid) * Nc + j0;
      bf16_t* dp = sV + tid * VSTR;
#pragma unroll
      for (int u = 0; u < (KT * 2) / 16; ++u)      // 8 x B128
        __builtin_amdgcn_global_load_async_to_lds_b128(
            (gv4i_t*)(hp + u * 8), (lv4i_t*)(dp + u * 8), 0, 0);
      // K tile: 64 keys x 32 qk-chans; 64 rows x 4 B128 chunks = 256 threads.
      int row = tid >> 2, ch = (tid & 3) * 8;
      const bf16_t* gp = gT + ((size_t)b * Nc + j0 + row) * CKc + ch;
      __builtin_amdgcn_global_load_async_to_lds_b128(
          (gv4i_t*)gp, (lv4i_t*)(sK + row * KSTR + ch), 0, 0);
    }
    __builtin_amdgcn_s_wait_asynccnt(0);  // my DMAs landed in LDS
    __syncthreads();                      // everyone's DMAs landed
#else
    // ---- synchronous staging fallback ----
    {
      int row = tid >> 2, ch = (tid & 3) * 8;
      const bf16_t* gp = gT + ((size_t)b * Nc + j0 + row) * CKc + ch;
      bf16_t* dp = sK + row * KSTR + ch;
#pragma unroll
      for (int u = 0; u < 8; ++u) dp[u] = gp[u];
    }
    {
      const bf16_t* hp = hB + ((size_t)b * Cc + tid) * Nc + j0;
      bf16_t* dp = sV + tid * VSTR;
#pragma unroll
      for (int u = 0; u < KT; ++u) dp[u] = hp[u];
    }
    __syncthreads();
    if (j0 + KT < Nc) {
      __builtin_prefetch(hB + ((size_t)b * Cc + tid) * Nc + j0 + KT, 0, 0);
      __builtin_prefetch(gT + ((size_t)b * Nc + j0 + KT + (tid & 31)) * CKc, 0, 0);
    }
#endif

    // ---- S = Q x K^T : four 16x16 tiles over 64 keys ----
    v8f s[4];
#pragma unroll
    for (int t = 0; t < 4; ++t) {
      v16bf bk;
#pragma unroll
      for (int e = 0; e < 16; ++e)                 // B layout: lane=N, K=e+16*half
        bk[e] = sK[(16 * t + l15) * KSTR + (e + 16 * half)];
      s[t] = __builtin_amdgcn_wmma_f32_16x16x32_bf16(false, aq, false, bk,
                                                     (short)0, zero8, false, false);
    }

    // ---- online softmax: row stats via 16-lane butterfly shuffles ----
    float scale[8];
#pragma unroll
    for (int r = 0; r < 8; ++r) {
      float mx = fmaxf(fmaxf(s[0][r], s[1][r]), fmaxf(s[2][r], s[3][r]));
#pragma unroll
      for (int m = 1; m < 16; m <<= 1) mx = fmaxf(mx, __shfl_xor(mx, m, 32));
      float mnew = fmaxf(mrow[r], mx);
      float sc = __expf(mrow[r] - mnew);
      float rs = 0.f;
#pragma unroll
      for (int t = 0; t < 4; ++t) {
        float p = __expf(s[t][r] - mnew);
        s[t][r] = p;
        rs += p;
      }
#pragma unroll
      for (int m = 1; m < 16; m <<= 1) rs += __shfl_xor(rs, m, 32);
      lrow[r] = lrow[r] * sc + rs;
      mrow[r] = mnew;
      scale[r] = sc;
    }

    // ---- P: C layout -> LDS [row][key] -> reload as two bf16 A fragments ----
    bf16_t* pw = sP + qg * (QPG * PSTR);
#pragma unroll
    for (int r = 0; r < 8; ++r) {
      int row = r + 8 * half;
#pragma unroll
      for (int t = 0; t < 4; ++t)
        pw[row * PSTR + 16 * t + l15] = (bf16_t)s[t][r];
    }
    v16bf ap0, ap1;
#pragma unroll
    for (int e = 0; e < 16; ++e) {
      int k = e + 16 * half;
      ap0[e] = pw[l15 * PSTR + k];                 // keys j0+ 0..31
      ap1[e] = pw[l15 * PSTR + 32 + k];            // keys j0+32..63
    }

    // ---- O = O*scale + P x V : 16 WMMAs (16q x 128ch per wave) ----
#pragma unroll
    for (int t = 0; t < 8; ++t) {
#pragma unroll
      for (int r = 0; r < 8; ++r) o[t][r] *= scale[r];
      v16bf bv;
      int cch = chalf * 128 + t * 16 + l15;        // B layout: lane=N (channel)
#pragma unroll
      for (int e = 0; e < 16; ++e)
        bv[e] = sV[cch * VSTR + (e + 16 * half)];
      o[t] = __builtin_amdgcn_wmma_f32_16x16x32_bf16(false, ap0, false, bv,
                                                     (short)0, o[t], false, false);
#pragma unroll
      for (int e = 0; e < 16; ++e)
        bv[e] = sV[cch * VSTR + 32 + (e + 16 * half)];
      o[t] = __builtin_amdgcn_wmma_f32_16x16x32_bf16(false, ap1, false, bv,
                                                     (short)0, o[t], false, false);
    }
  }

  // ---- epilogue: normalize, gamma*out + x ----
  float g = gamma[0];
#pragma unroll
  for (int r = 0; r < 8; ++r) lrow[r] = 1.f / lrow[r];
#pragma unroll
  for (int t = 0; t < 8; ++t) {
    int cch = chalf * 128 + t * 16 + l15;
#pragma unroll
    for (int r = 0; r < 8; ++r) {
      int i = q0 + r + 8 * half;
      size_t idx = ((size_t)b * Cc + cch) * (size_t)Nc + i;
      out[idx] = g * (o[t][r] * lrow[r]) + x[idx];
    }
  }
}

// ---------------------------------------------------------------------------
extern "C" void kernel_launch(void* const* d_in, const int* in_sizes, int n_in,
                              void* d_out, int out_size, void* d_ws, size_t ws_size,
                              hipStream_t stream) {
  const float* x     = (const float*)d_in[0];
  const float* Wf    = (const float*)d_in[1];
  const float* bf    = (const float*)d_in[2];
  const float* Wg    = (const float*)d_in[3];
  const float* bg    = (const float*)d_in[4];
  const float* Wh    = (const float*)d_in[5];
  const float* bh    = (const float*)d_in[6];
  const float* gamma = (const float*)d_in[7];
  float* out = (float*)d_out;

  char* ws = (char*)d_ws;
  size_t fgBytes = (size_t)Bc * Nc * CKc * sizeof(bf16_t);   // 1 MB each
  bf16_t* fT = (bf16_t*)ws;
  bf16_t* gT = (bf16_t*)(ws + fgBytes);
  bf16_t* hB = (bf16_t*)(ws + 2 * fgBytes);                  // 8 MB

  proj_fg<<<(Bc * Nc) / 256, 256, 0, stream>>>(x, Wf, bf, Wg, bg, fT, gT);
  proj_h<<<Bc * Cc * (Nc / 256), 256, 0, stream>>>(x, Wh, bh, hB);
  attn<<<Bc * (Nc / (GROUPS * QPG)), 256, 0, stream>>>(fT, gT, hB, x, gamma, out);
}
